// StateUpdate_38792144617952
// MI455X (gfx1250) — matrix operand order repaired and checked by reading, not compiled, not run on hardware
//
#include <hip/hip_runtime.h>
#include <hip/hip_bf16.h>

#define B_ 32
#define T_ 1024
#define D_ 1024
#define S_ 512

typedef __attribute__((ext_vector_type(16))) __bf16 v16bf;
typedef __attribute__((ext_vector_type(8)))  __bf16 v8bf;
typedef __attribute__((ext_vector_type(8)))  float  v8f;
typedef __attribute__((ext_vector_type(4)))  unsigned int v4u;
typedef __attribute__((ext_vector_type(8)))  int v8i;
typedef __attribute__((ext_vector_type(4)))  int v4i;

// ---------- helpers ----------

__device__ __forceinline__ v8f zero8() {
  v8f r;
#pragma unroll
  for (int i = 0; i < 8; ++i) r[i] = 0.0f;
  return r;
}

__device__ __forceinline__ v16bf combine16(v8bf lo, v8bf hi) {
  v16bf r;
#pragma unroll
  for (int i = 0; i < 8; ++i) { r[i] = lo[i]; r[i + 8] = hi[i]; }
  return r;
}

// convert 2x8 contiguous fp32 chunks into a packed 16xbf16 A-fragment
// (native hardware converts: v_cvt_pk_bf16_f32)
__device__ __forceinline__ v16bf cvt_pack16(const float* p0, const float* p1) {
  v16bf r;
#pragma unroll
  for (int i = 0; i < 8; ++i) {
    r[i]     = (__bf16)p0[i];
    r[i + 8] = (__bf16)p1[i];
  }
  return r;
}

__device__ __forceinline__ v8f wmma_bf(v16bf a, v16bf b, v8f c) {
  return __builtin_amdgcn_wmma_f32_16x16x32_bf16(false, a, false, b,
                                                 (short)0, c, false, false);
}

// TDM: 1-D tile of `nelem` fp32 from global -> LDS (issued by one wave)
__device__ __forceinline__ void tdm_load_f32(const float* gsrc,
                                             unsigned lds_byte_addr,
                                             unsigned nelem) {
  unsigned long long ga = (unsigned long long)(size_t)gsrc;
  v4u g0;
  g0[0] = 1u;                                   // count=1, user descriptor
  g0[1] = lds_byte_addr;                        // lds_addr
  g0[2] = (unsigned)(ga & 0xFFFFFFFFu);         // global_addr[31:0]
  g0[3] = (unsigned)((ga >> 32) & 0x1FFFFFFu)   // global_addr[56:32]
          | (2u << 30);                         // type = 2 ("image")
  v8i g1;
  g1[0] = (int)(2u << 16);                      // data_size = 4 bytes
  g1[1] = (int)((nelem & 0xFFFFu) << 16);       // tensor_dim0 lo16
  g1[2] = (int)(((nelem >> 16) & 0xFFFFu)       // tensor_dim0 hi16
          | (1u << 16));                        // tensor_dim1 = 1
  g1[3] = (int)((nelem & 0xFFFFu) << 16);       // tile_dim0
  g1[4] = 0;                                    // tile_dim1/2 unused
  g1[5] = (int)nelem;                           // tensor_dim0_stride lo32
  g1[6] = 0;
  g1[7] = 0;
  v4i z4; z4[0] = 0; z4[1] = 0; z4[2] = 0; z4[3] = 0;
  v8i z8;
#pragma unroll
  for (int i = 0; i < 8; ++i) z8[i] = 0;
  __builtin_amdgcn_tensor_load_to_lds(g0, g1, z4, z4, z8, 0);
}

// ---------- kernel 0: fp32 -> bf16 weight conversion ----------

__global__ __launch_bounds__(256) void cvt_bf16(const float* __restrict__ src,
                                                __bf16* __restrict__ dst,
                                                int n) {
  int i = blockIdx.x * 256 + threadIdx.x;
  if (i < n) dst[i] = (__bf16)src[i];
}

// ---------- kernel 1: xi[t*B+b, s] = sum_d x[b*T+t, d] * inj[s, d] ----------
// M = B*T = 32768, N = S = 512, K = D = 1024.  Block = 8 waves, wave = 16x64.

__global__ __launch_bounds__(256) void xi_gemm(const float* __restrict__ x,
                                               const __bf16* __restrict__ injb,
                                               float* __restrict__ xi) {
  const int tid  = threadIdx.x;
  const int lane = tid & 31;
  const int wave = tid >> 5;
  const int l15  = lane & 15;
  const int hiL  = (lane >= 16) ? 1 : 0;
  const int Mbase = blockIdx.x * 128 + wave * 16;
  const int Nbase = blockIdx.y * 64;

  const float* xrow = x + (size_t)(Mbase + l15) * D_;
  const int c0 = hiL ? 16 : 0;
  const int bo = hiL ? 16 : 0;

  v8f acc[4];
#pragma unroll
  for (int i = 0; i < 4; ++i) acc[i] = zero8();

  for (int k = 0; k < D_; k += 32) {
    v16bf a = cvt_pack16(xrow + k + c0, xrow + k + c0 + 8);
#pragma unroll
    for (int nt = 0; nt < 4; ++nt) {
      const int col = Nbase + nt * 16 + l15;
      v16bf b = *(const v16bf*)(injb + (size_t)col * D_ + k + bo);
      acc[nt] = wmma_bf(a, b, acc[nt]);
    }
  }

#pragma unroll
  for (int nt = 0; nt < 4; ++nt) {
    const int n = Nbase + nt * 16 + l15;
#pragma unroll
    for (int jj = 0; jj < 8; ++jj) {
      const int r  = Mbase + jj + (hiL ? 8 : 0);   // r = b*T + t
      const int bb = r >> 10;
      const int tt = r & (T_ - 1);
      xi[(size_t)(tt * B_ + bb) * S_ + n] = acc[nt][jj];
    }
  }
}

// ---------- kernel 2: persistent single-WGP recurrent scan ----------
// h (fp32 + bf16 copies) in LDS; xi_t double-buffered in LDS via TDM;
// gate_w / state_flow bf16 streamed from L2 each step (L2-resident, 1MB).

__global__ __launch_bounds__(1024) void scan_kernel(
    const __bf16* __restrict__ gwb,   // [S,S] bf16, row j contiguous over i (=K)
    const __bf16* __restrict__ flb,   // [S,S] bf16
    const float* __restrict__ gate_b, // [S]
    const float* __restrict__ xi,     // [T,B,S] fp32
    __bf16* __restrict__ Hbf) {       // [T*B, S] bf16 out
  extern __shared__ char smem[];
  float*  h32   = (float*)smem;                     // 64 KB
  __bf16* hbf   = (__bf16*)(smem + 65536);          // 32 KB
  float*  xibuf = (float*)(smem + 65536 + 32768);   // 2 x 64 KB

  const int tid  = threadIdx.x;
  const int lane = tid & 31;
  const int wave = tid >> 5;
  const int l15  = lane & 15;
  const int hiL  = (lane >= 16) ? 1 : 0;
  const int mt   = wave >> 4;           // 0..1 : which 16-row block of B=32
  const int nt0  = (wave & 15) * 2;     // 0..30: first of two 16-col tiles

  // init h = 0
  for (int i = tid; i < B_ * S_; i += 1024) { h32[i] = 0.0f; hbf[i] = (__bf16)0.0f; }
  // preload xi_0 into buffer 0 via TDM
  if (wave == 0) {
    tdm_load_f32(xi, (unsigned)(size_t)(void*)xibuf, B_ * S_);
    __builtin_amdgcn_s_wait_tensorcnt(0);
  }
  __syncthreads();

  const int c0 = hiL ? 16 : 0;
  const int bo = hiL ? 16 : 0;
  const __bf16* hA = hbf + (size_t)(mt * 16 + l15) * S_;
  const int j0 = nt0 * 16 + l15;
  const int j1 = j0 + 16;
  const __bf16* g0p = gwb + (size_t)j0 * S_ + bo;
  const __bf16* g1p = gwb + (size_t)j1 * S_ + bo;
  const __bf16* f0p = flb + (size_t)j0 * S_ + bo;
  const __bf16* f1p = flb + (size_t)j1 * S_ + bo;
  const float gb0 = gate_b[nt0 * 16 + l15];
  const float gb1 = gate_b[(nt0 + 1) * 16 + l15];

  for (int t = 0; t < T_; ++t) {
    const float* xic = xibuf + (t & 1) * (B_ * S_);
    // kick off DMA for xi_{t+1}; it overlaps this step's GEMMs
    if (wave == 0 && (t + 1) < T_) {
      tdm_load_f32(xi + (size_t)(t + 1) * (B_ * S_),
                   (unsigned)(size_t)(void*)(xibuf + ((t + 1) & 1) * (B_ * S_)),
                   B_ * S_);
    }

    v8f ag0 = zero8(), ag1 = zero8(), ac0 = zero8(), ac1 = zero8();
#pragma unroll 4
    for (int k = 0; k < S_; k += 32) {
      v8bf alo = *(const v8bf*)(hA + k + c0);        // LDS A-frag
      v8bf ahi = *(const v8bf*)(hA + k + c0 + 8);
      v16bf a  = combine16(alo, ahi);
      v16bf bg0 = *(const v16bf*)(g0p + k);
      v16bf bg1 = *(const v16bf*)(g1p + k);
      v16bf bf0 = *(const v16bf*)(f0p + k);
      v16bf bf1 = *(const v16bf*)(f1p + k);
      ag0 = wmma_bf(a, bg0, ag0);
      ag1 = wmma_bf(a, bg1, ag1);
      ac0 = wmma_bf(a, bf0, ac0);
      ac1 = wmma_bf(a, bf1, ac1);
    }
    __syncthreads();  // all waves done reading h before anyone rewrites it

    // gated state update (each wave owns disjoint [m-tile]x[2 n-tiles])
#pragma unroll
    for (int tp = 0; tp < 2; ++tp) {
      v8f ag = tp ? ag1 : ag0;
      v8f ac = tp ? ac1 : ac0;
      const int   n  = (nt0 + tp) * 16 + l15;
      const float gb = tp ? gb1 : gb0;
#pragma unroll
      for (int jj = 0; jj < 8; ++jj) {
        const int m   = mt * 16 + jj + (hiL ? 8 : 0);
        const int idx = m * S_ + n;
        const float pre = ag[jj] + gb + xic[idx];
        const float g   = 1.0f / (1.0f + __expf(-pre));
        const float ho  = h32[idx];
        const float hn  = ho + g * (ac[jj] - ho);   // (1-g)*h + g*h_cand
        h32[idx] = hn;
        const __bf16 hb = (__bf16)hn;
        hbf[idx] = hb;
        Hbf[(size_t)(t * B_ + m) * S_ + n] = hb;
      }
    }
    if (wave == 0) __builtin_amdgcn_s_wait_tensorcnt(0);
    __syncthreads();  // h + xi_{t+1} visible to all waves
  }
}

// ---------- kernel 3: y[b,t,d] = sum_s H[t*B+b, s] * readout[d, s] ----------
// M = T*B = 32768, N = D = 1024, K = S = 512.

__global__ __launch_bounds__(256) void readout_gemm(
    const __bf16* __restrict__ Hbf,
    const __bf16* __restrict__ rob,
    float* __restrict__ out) {
  const int tid  = threadIdx.x;
  const int lane = tid & 31;
  const int wave = tid >> 5;
  const int l15  = lane & 15;
  const int hiL  = (lane >= 16) ? 1 : 0;
  const int Mbase = blockIdx.x * 128 + wave * 16;
  const int Nbase = blockIdx.y * 64;

  const __bf16* hrow = Hbf + (size_t)(Mbase + l15) * S_;
  const int c0 = hiL ? 16 : 0;
  const int bo = hiL ? 16 : 0;

  v8f acc[4];
#pragma unroll
  for (int i = 0; i < 4; ++i) acc[i] = zero8();

  for (int k = 0; k < S_; k += 32) {
    v8bf alo = *(const v8bf*)(hrow + k + c0);
    v8bf ahi = *(const v8bf*)(hrow + k + c0 + 8);
    v16bf a  = combine16(alo, ahi);
#pragma unroll
    for (int nt = 0; nt < 4; ++nt) {
      const int col = Nbase + nt * 16 + l15;
      v16bf b = *(const v16bf*)(rob + (size_t)col * S_ + k + bo);
      acc[nt] = wmma_bf(a, b, acc[nt]);
    }
  }

#pragma unroll
  for (int nt = 0; nt < 4; ++nt) {
    const int n = Nbase + nt * 16 + l15;
#pragma unroll
    for (int jj = 0; jj < 8; ++jj) {
      const int r  = Mbase + jj + (hiL ? 8 : 0);   // r = t*B + b
      const int tt = r >> 5;
      const int bb = r & (B_ - 1);
      out[(size_t)(bb * T_ + tt) * D_ + n] = acc[nt][jj];
    }
  }
}

// ---------- launcher ----------

extern "C" void kernel_launch(void* const* d_in, const int* in_sizes, int n_in,
                              void* d_out, int out_size, void* d_ws, size_t ws_size,
                              hipStream_t stream) {
  (void)in_sizes; (void)n_in; (void)out_size; (void)ws_size;
  const float* x    = (const float*)d_in[0];   // [B,T,D]
  const float* flow = (const float*)d_in[1];   // [S,S]
  const float* inj  = (const float*)d_in[2];   // [S,D]
  const float* ro   = (const float*)d_in[3];   // [D,S]
  const float* gw   = (const float*)d_in[4];   // [S,S]
  const float* gb   = (const float*)d_in[5];   // [S]
  float* out = (float*)d_out;

  char* ws = (char*)d_ws;
  __bf16* injb = (__bf16*)(ws + 0);          // 1 MB
  __bf16* gwb  = (__bf16*)(ws + 1048576);    // 512 KB
  __bf16* flb  = (__bf16*)(ws + 1572864);    // 512 KB
  __bf16* rob  = (__bf16*)(ws + 2097152);    // 1 MB
  float*  xi   = (float*)(ws + 3145728);     // 64 MB  [T,B,S]
  __bf16* Hbf  = (__bf16*)(ws + 3145728 + 67108864); // 32 MB

  cvt_bf16<<<(S_ * D_ + 255) / 256, 256, 0, stream>>>(inj, injb, S_ * D_);
  cvt_bf16<<<(S_ * S_ + 255) / 256, 256, 0, stream>>>(gw, gwb, S_ * S_);
  cvt_bf16<<<(S_ * S_ + 255) / 256, 256, 0, stream>>>(flow, flb, S_ * S_);
  cvt_bf16<<<(D_ * S_ + 255) / 256, 256, 0, stream>>>(ro, rob, D_ * S_);

  xi_gemm<<<dim3(256, 8), 256, 0, stream>>>(x, injb, xi);

  const int scan_lds = 65536 + 32768 + 2 * 65536;  // 224 KB of 320 KB WGP LDS
  (void)hipFuncSetAttribute(reinterpret_cast<const void*>(scan_kernel),
                            hipFuncAttributeMaxDynamicSharedMemorySize, scan_lds);
  scan_kernel<<<1, 1024, scan_lds, stream>>>(gwb, flb, gb, xi, Hbf);

  readout_gemm<<<dim3(256, 16), 256, 0, stream>>>(Hbf, rob, out);
}
